// MoFo_Backbone_4913442587138
// MI455X (gfx1250) — matrix operand order repaired and checked by reading, not compiled, not run on hardware
//
#include <hip/hip_runtime.h>

#define B_      256
#define CYCLE_  288
#define DIM_    512
#define HEAD_   8
#define HD_     64
#define M_TOT   (B_ * CYCLE_)   // 73728

// ---------------- WMMA fragment types (gfx1250, wave32) ----------------
typedef __attribute__((ext_vector_type(16))) __bf16         v16bf;
typedef __attribute__((ext_vector_type(16))) unsigned short v16us;
typedef __attribute__((ext_vector_type(8)))  unsigned short v8us;
typedef __attribute__((ext_vector_type(8)))  float          v8f;
typedef __attribute__((ext_vector_type(4)))  unsigned int   v4u;
typedef __attribute__((ext_vector_type(4)))  int            v4i;
typedef __attribute__((ext_vector_type(8)))  int            v8i;

union FragBF {
  v16us u;
  v16bf b;
  v8us  h[2];   // h[0] = elements 0..7, h[1] = 8..15
};

__device__ __forceinline__ unsigned short f2bf(float f) {
  unsigned u = __float_as_uint(f);
  u += 0x7fffu + ((u >> 16) & 1u);          // round-to-nearest-even
  return (unsigned short)(u >> 16);
}
__device__ __forceinline__ v8f zero8() {
  v8f z = {0.f,0.f,0.f,0.f,0.f,0.f,0.f,0.f};
  return z;
}
__device__ __forceinline__ unsigned lds_off32(const void* p) {
  return (unsigned)(size_t)p;               // low 32 bits = LDS byte offset
}

// ---- Tensor Data Mover: 2D tile (bf16) -> LDS, with LDS row padding ----
// D# packing per CDNA5 ISA Sec 8.3/8.4. tensor_dim ==
// tile_dim (tiles always in-bounds). pad codes: interval 0=2dw..4=32dw;
// amount = code+1 dwords. 6-arg builtin form (clang-23 / therock-10.0):
// (uint32x4 g0, int32x8 g1, int32x4 g2, int32x4 g3, int32x8 g4, i32 cpol).
__device__ __forceinline__ void tdm_load_2d(const void* gaddr, unsigned ldsoff,
                                            unsigned tile_w, unsigned tile_h,
                                            unsigned row_stride_elems,
                                            unsigned pad_interval_code,
                                            unsigned pad_amount_code) {
  unsigned long long ga = (unsigned long long)(size_t)gaddr;
  v4u g0;
  g0[0] = 1u;                                            // count=1, user mode
  g0[1] = ldsoff;                                        // lds_addr
  g0[2] = (unsigned)ga;                                  // global_addr[31:0]
  g0[3] = (unsigned)((ga >> 32) & 0x01ffffffull)         // global_addr[56:32]
        | (2u << 30);                                    // type = 2 (image)
  v8i g1;
  g1[0] = (int)((1u << 16)                               // data_size = 2B
              | (1u << 20)                               // pad_enable
              | (pad_interval_code << 22)
              | (pad_amount_code << 25));
  g1[1] = (int)((tile_w & 0xffffu) << 16);               // tensor_dim0[15:0]
  g1[2] = (int)(((tile_w >> 16) & 0xffffu)               // tensor_dim0[31:16]
              | ((tile_h & 0xffffu) << 16));             // tensor_dim1[15:0]
  g1[3] = (int)(((tile_h >> 16) & 0xffffu)               // tensor_dim1[31:16]
              | ((tile_w & 0xffffu) << 16));             // tile_dim0
  g1[4] = (int)(tile_h & 0xffffu);                       // tile_dim1 (dim2=0)
  g1[5] = (int)row_stride_elems;                         // dim0_stride[31:0]
  g1[6] = 0;                                             // stride hi / dim1_stride lo
  g1[7] = 0;
  v4i z4 = {0, 0, 0, 0};
  v8i z8 = {0, 0, 0, 0, 0, 0, 0, 0};
  __builtin_amdgcn_tensor_load_to_lds(g0, g1, z4, z4, z8, 0);
}

// ---------------- f32 -> bf16 weight conversion ----------------
__global__ void convert_bf16_kernel(const float* __restrict__ in,
                                    unsigned short* __restrict__ outp, int n) {
  int i = blockIdx.x * blockDim.x + threadIdx.x;
  if (i < n) outp[i] = f2bf(in[i]);
}

// ---------------- decay-bias log table ----------------
__global__ __launch_bounds__(288)
void decay_kernel(const float* __restrict__ a1, const float* __restrict__ a2,
                  const float* __restrict__ bp1, const float* __restrict__ bp2,
                  float* __restrict__ logbias) {
  int t = threadIdx.x;
  float p1 = a1[t] * a2[t];
  float p2 = bp1[t] * bp2[t];
  for (int off = 16; off; off >>= 1) {
    p1 += __shfl_xor(p1, off, 32);
    p2 += __shfl_xor(p2, off, 32);
  }
  __shared__ float q1[9], q2[9];
  if ((t & 31) == 0) { q1[t >> 5] = p1; q2[t >> 5] = p2; }
  __syncthreads();
  float d1 = 0.f, d2 = 0.f;
  #pragma unroll
  for (int i = 0; i < 9; ++i) { d1 += q1[i]; d2 += q2[i]; }
  float a  = 1.f / (1.f + __expf(-d1));
  float bb = (1.f / (1.f + __expf(-d2))) * (float)CYCLE_;
  float dist = (float)t;
  float diff = fminf(dist, (float)CYCLE_ - dist);
  float val  = 1.f / (1.f + __expf(a * (diff - bb)))
             + __expf(-diff) / (1.f + __expf(a * bb));
  logbias[t] = __logf(val);
}

// ---------------- RMSNorm (512-wide rows) -> bf16 ----------------
__global__ __launch_bounds__(256)
void rmsnorm_kernel(const float* __restrict__ x, const float* __restrict__ scale,
                    const float* __restrict__ offset, unsigned short* __restrict__ outp) {
  const int row = blockIdx.x;
  const int t   = threadIdx.x;
  const float* xr = x + (size_t)row * DIM_;
  float v0 = xr[t], v1 = xr[t + 256];
  float ss = v0 * v0 + v1 * v1;
  for (int off = 16; off; off >>= 1) ss += __shfl_xor(ss, off, 32);
  __shared__ float partial[8];
  if ((t & 31) == 0) partial[t >> 5] = ss;
  __syncthreads();
  float tot = 0.f;
  #pragma unroll
  for (int i = 0; i < 8; ++i) tot += partial[i];
  float rms = sqrtf(tot * (1.0f / (float)DIM_));
  float inv = 1.0f / (rms + 1e-8f);
  outp[(size_t)row * DIM_ + t]       = f2bf(scale[t] * (v0 * inv) + offset[t]);
  outp[(size_t)row * DIM_ + t + 256] = f2bf(scale[t + 256] * (v1 * inv) + offset[t + 256]);
}

// ---------------- bf16 WMMA GEMM: C = A@W + bias (+res) ----------------
// Block tile 128(M) x 64(N), K-step 32, 8 waves. A tile via TDM (row = 64B =
// 16 dwords -> pad_interval 3; pad 4 dwords -> code 3 => LDS stride 40 elems).
// W staged transposed (manual; TDM cannot transpose). All fragments are
// contiguous 16B runs -> ds_load_b128 pairs.
#define GBM 128
#define GBN 64
#define GBK 32
#define GLDA 40
#define GLDBT 40

template <int OUT_F32, int ADD_RES>
__global__ __launch_bounds__(256)
void gemm_bf16_kernel(const unsigned short* __restrict__ A,
                      const unsigned short* __restrict__ W,
                      const float* __restrict__ bias,
                      const float* __restrict__ res,
                      unsigned short* __restrict__ Obf,
                      float* __restrict__ Of,
                      int M, int N, int K) {
  __shared__ unsigned short sA[GBM * GLDA];
  __shared__ unsigned short sBt[GBN * GLDBT];
  const int t  = threadIdx.x;
  const int m0 = blockIdx.y * GBM;
  const int n0 = blockIdx.x * GBN;
  const int w = t >> 5, lane = t & 31, lm = lane & 15, hi = lane >> 4;

  v8f acc[4];
  #pragma unroll
  for (int i = 0; i < 4; ++i) acc[i] = zero8();

  for (int k0 = 0; k0 < K; k0 += GBK) {
    // A tile 128x32 via Tensor Data Mover (async DMA -> LDS)
    if (w == 0) {
      tdm_load_2d(A + (size_t)m0 * K + k0, lds_off32(sA),
                  GBK, GBM, (unsigned)K, 3, 3);
    }
    // W tile 32x64 -> transposed LDS sBt[n][k]
    #pragma unroll
    for (int i = 0; i < 2; ++i) {
      int vi  = t + i * 256;            // 0..511
      int row = vi >> 4;                // k: 0..31
      int c4  = (vi & 15) * 4;          // n
      union { uint2 q; unsigned short e[4]; } u;
      u.q = *(const uint2*)(W + (size_t)(k0 + row) * N + n0 + c4);
      #pragma unroll
      for (int j = 0; j < 4; ++j) sBt[(c4 + j) * GLDBT + row] = u.e[j];
    }
    if (k0 + GBK < K && t < GBK)
      __builtin_prefetch(W + (size_t)(k0 + GBK + t) * N + n0, 0, 0);
    if (w == 0) __builtin_amdgcn_s_wait_tensorcnt(0);
    __syncthreads();

    FragBF a;
    const int arow = w * 16 + lm;
    a.h[0] = *(const v8us*)(sA + arow * GLDA + hi * 16);
    a.h[1] = *(const v8us*)(sA + arow * GLDA + hi * 16 + 8);
    #pragma unroll
    for (int nt = 0; nt < 4; ++nt) {
      FragBF bfr;
      const unsigned short* bp = sBt + (nt * 16 + lm) * GLDBT + hi * 16;
      bfr.h[0] = *(const v8us*)(bp);
      bfr.h[1] = *(const v8us*)(bp + 8);
      acc[nt] = __builtin_amdgcn_wmma_f32_16x16x32_bf16(
          false, a.b, false, bfr.b, (short)0, acc[nt], false, false);
    }
    __syncthreads();
  }

  const int mb = m0 + w * 16 + hi * 8;
  #pragma unroll
  for (int nt = 0; nt < 4; ++nt) {
    int col  = n0 + nt * 16 + lm;
    float bv = bias[col];
    #pragma unroll
    for (int r = 0; r < 8; ++r) {
      size_t idx = (size_t)(mb + r) * N + col;
      float v = acc[nt][r] + bv;
      if (OUT_F32) {
        if (ADD_RES) v += res[idx];
        Of[idx] = v;
      } else {
        Obf[idx] = f2bf(v);
      }
    }
  }
}

// ---------------- fused dual GEMM + SwiGLU: O = silu(A@W1+b1)*(A@W2+b2) ----
__global__ __launch_bounds__(256)
void gemm_swiglu_kernel(const unsigned short* __restrict__ A,
                        const unsigned short* __restrict__ W1,
                        const float* __restrict__ b1,
                        const unsigned short* __restrict__ W2,
                        const float* __restrict__ b2,
                        unsigned short* __restrict__ O,
                        int M, int N, int K) {
  __shared__ unsigned short sA[GBM * GLDA];
  __shared__ unsigned short sB1t[GBN * GLDBT];
  __shared__ unsigned short sB2t[GBN * GLDBT];
  const int t  = threadIdx.x;
  const int m0 = blockIdx.y * GBM;
  const int n0 = blockIdx.x * GBN;
  const int w = t >> 5, lane = t & 31, lm = lane & 15, hi = lane >> 4;

  v8f ac1[4], ac2[4];
  #pragma unroll
  for (int i = 0; i < 4; ++i) { ac1[i] = zero8(); ac2[i] = zero8(); }

  for (int k0 = 0; k0 < K; k0 += GBK) {
    if (w == 0) {
      tdm_load_2d(A + (size_t)m0 * K + k0, lds_off32(sA),
                  GBK, GBM, (unsigned)K, 3, 3);
    }
    #pragma unroll
    for (int i = 0; i < 2; ++i) {
      int vi = t + i * 256, row = vi >> 4, c4 = (vi & 15) * 4;
      union { uint2 q; unsigned short e[4]; } u1, u2;
      u1.q = *(const uint2*)(W1 + (size_t)(k0 + row) * N + n0 + c4);
      u2.q = *(const uint2*)(W2 + (size_t)(k0 + row) * N + n0 + c4);
      #pragma unroll
      for (int j = 0; j < 4; ++j) {
        sB1t[(c4 + j) * GLDBT + row] = u1.e[j];
        sB2t[(c4 + j) * GLDBT + row] = u2.e[j];
      }
    }
    if (k0 + GBK < K && t < GBK) {
      __builtin_prefetch(W1 + (size_t)(k0 + GBK + t) * N + n0, 0, 0);
      __builtin_prefetch(W2 + (size_t)(k0 + GBK + t) * N + n0, 0, 0);
    }
    if (w == 0) __builtin_amdgcn_s_wait_tensorcnt(0);
    __syncthreads();

    FragBF a;
    const int arow = w * 16 + lm;
    a.h[0] = *(const v8us*)(sA + arow * GLDA + hi * 16);
    a.h[1] = *(const v8us*)(sA + arow * GLDA + hi * 16 + 8);
    #pragma unroll
    for (int nt = 0; nt < 4; ++nt) {
      FragBF f1, f2;
      const unsigned short* p1 = sB1t + (nt * 16 + lm) * GLDBT + hi * 16;
      const unsigned short* p2 = sB2t + (nt * 16 + lm) * GLDBT + hi * 16;
      f1.h[0] = *(const v8us*)(p1);
      f1.h[1] = *(const v8us*)(p1 + 8);
      f2.h[0] = *(const v8us*)(p2);
      f2.h[1] = *(const v8us*)(p2 + 8);
      ac1[nt] = __builtin_amdgcn_wmma_f32_16x16x32_bf16(
          false, a.b, false, f1.b, (short)0, ac1[nt], false, false);
      ac2[nt] = __builtin_amdgcn_wmma_f32_16x16x32_bf16(
          false, a.b, false, f2.b, (short)0, ac2[nt], false, false);
    }
    __syncthreads();
  }

  const int mb = m0 + w * 16 + hi * 8;
  #pragma unroll
  for (int nt = 0; nt < 4; ++nt) {
    int col   = n0 + nt * 16 + lm;
    float bv1 = b1[col], bv2 = b2[col];
    #pragma unroll
    for (int r = 0; r < 8; ++r) {
      size_t idx = (size_t)(mb + r) * N + col;
      float u  = ac1[nt][r] + bv1;
      float vv = ac2[nt][r] + bv2;
      float s  = u / (1.f + __expf(-u));      // silu
      O[idx] = f2bf(s * vv);
    }
  }
}

// ---------------- attention: per (b, head, 16-query-row tile) ----------------
// 9 waves (288 thr). K and Q tiles arrive via TDM (row 128B = 32 dw -> pad
// interval code 4, pad 4 dw -> stride 72 elems). V staged transposed manually.
#define SKS 72    // sK/sQ row stride (elems): 144B rows
#define SVS 296   // sVt/sP row stride: 592B rows
#define SSTR 292  // sS row stride (floats)
#define ATTN_LDS_BYTES (288*SKS*2 + 64*SVS*2 + 16*SKS*2 + 16*SSTR*4 + 16*SVS*2)

__global__ __launch_bounds__(288)
void attn_kernel(const unsigned short* __restrict__ qkv,    // [B*L, 1536] bf16
                 const float* __restrict__ logbias,          // [288]
                 unsigned short* __restrict__ ao) {          // [B*L, 512] bf16
  extern __shared__ char smem[];
  unsigned short* sK  = (unsigned short*)smem;            // [288][SKS]
  unsigned short* sVt = sK + 288 * SKS;                   // [64][SVS]
  unsigned short* sQ  = sVt + 64 * SVS;                   // [16][SKS]
  float*          sS  = (float*)(sQ + 16 * SKS);          // [16][SSTR]
  unsigned short* sP  = (unsigned short*)(sS + 16 * SSTR);// [16][SVS]

  const int bid = blockIdx.x;
  const int qt  = bid % 18;
  const int hh  = (bid / 18) % HEAD_;
  const int b   = bid / (18 * HEAD_);
  const int t   = threadIdx.x;
  const int w = t >> 5, lane = t & 31, lm = lane & 15, hi = lane >> 4;

  const unsigned short* base = qkv + (size_t)b * CYCLE_ * 1536;

  // K tile 288x64 and Q tile 16x64 via Tensor Data Mover
  if (w == 0) {
    tdm_load_2d(base + hh * 192 + 64, lds_off32(sK), 64, 288, 1536, 4, 3);
    tdm_load_2d(base + (size_t)(qt * 16) * 1536 + hh * 192, lds_off32(sQ),
                64, 16, 1536, 4, 3);
  }
  // V staged transposed (TDM cannot transpose): sVt[d][l]
  for (int g = t; g < 288 * 8; g += 288) {
    int l = g >> 3, d8 = (g & 7) * 8;
    union { uint4 q; unsigned short e[8]; } vv;
    vv.q = *(const uint4*)(base + (size_t)l * 1536 + hh * 192 + 128 + d8);
    #pragma unroll
    for (int j = 0; j < 8; ++j) sVt[(d8 + j) * SVS + l] = vv.e[j];
  }
  if (w == 0) __builtin_amdgcn_s_wait_tensorcnt(0);
  __syncthreads();

  // ---- scores = (Q Kt)*scale + logbias : 18 column tiles over 9 waves ----
  #pragma unroll
  for (int cc = 0; cc < 2; ++cc) {
    int ct = w * 2 + cc;                       // 0..17
    v8f acc = zero8();
    #pragma unroll
    for (int ks = 0; ks < 2; ++ks) {           // K = 64 in two 32-steps
      FragBF a, bfr;
      a.h[0] = *(const v8us*)(sQ + lm * SKS + ks * 32 + hi * 16);
      a.h[1] = *(const v8us*)(sQ + lm * SKS + ks * 32 + hi * 16 + 8);
      const unsigned short* kp = sK + (ct * 16 + lm) * SKS + ks * 32 + hi * 16;
      bfr.h[0] = *(const v8us*)(kp);
      bfr.h[1] = *(const v8us*)(kp + 8);
      acc = __builtin_amdgcn_wmma_f32_16x16x32_bf16(
          false, a.b, false, bfr.b, (short)0, acc, false, false);
    }
    #pragma unroll
    for (int r = 0; r < 8; ++r) {
      int m = hi * 8 + r;
      int c = ct * 16 + lm;
      int l = qt * 16 + m;
      int dist = l - c; if (dist < 0) dist = -dist;
      sS[m * SSTR + c] = acc[r] * 0.125f + logbias[dist];
    }
  }
  __syncthreads();

  // ---- softmax over 288 cols: wave w handles rows 2w, 2w+1 (w<8) ----
  if (w < 8) {
    #pragma unroll
    for (int rr = 0; rr < 2; ++rr) {
      int m = w * 2 + rr;
      float mx = -1e30f;
      for (int c = lane; c < 288; c += 32) mx = fmaxf(mx, sS[m * SSTR + c]);
      for (int off = 16; off; off >>= 1) mx = fmaxf(mx, __shfl_xor(mx, off, 32));
      float sum = 0.f;
      for (int c = lane; c < 288; c += 32) {
        float e = __expf(sS[m * SSTR + c] - mx);
        sS[m * SSTR + c] = e;
        sum += e;
      }
      for (int off = 16; off; off >>= 1) sum += __shfl_xor(sum, off, 32);
      float inv = 1.f / sum;
      for (int c = lane; c < 288; c += 32) sP[m * SVS + c] = f2bf(sS[m * SSTR + c] * inv);
    }
  }
  __syncthreads();

  // ---- out = P @ V : waves 0..3 own 16-wide d-tiles, 9 K-steps of 32 ----
  if (w < 4) {
    int nt = w;
    v8f acc = zero8();
    for (int kk = 0; kk < 9; ++kk) {
      FragBF a, bfr;
      a.h[0] = *(const v8us*)(sP + lm * SVS + kk * 32 + hi * 16);
      a.h[1] = *(const v8us*)(sP + lm * SVS + kk * 32 + hi * 16 + 8);
      const unsigned short* vp = sVt + (nt * 16 + lm) * SVS + kk * 32 + hi * 16;
      bfr.h[0] = *(const v8us*)(vp);
      bfr.h[1] = *(const v8us*)(vp + 8);
      acc = __builtin_amdgcn_wmma_f32_16x16x32_bf16(
          false, a.b, false, bfr.b, (short)0, acc, false, false);
    }
    #pragma unroll
    for (int r = 0; r < 8; ++r) {
      int m = hi * 8 + r;
      int l = qt * 16 + m;
      int d = nt * 16 + lm;
      ao[((size_t)b * CYCLE_ + l) * DIM_ + hh * 64 + d] = f2bf(acc[r]);
    }
  }
}

// ---------------- host orchestration ----------------
extern "C" void kernel_launch(void* const* d_in, const int* in_sizes, int n_in,
                              void* d_out, int out_size, void* d_ws, size_t ws_size,
                              hipStream_t stream) {
  (void)in_sizes; (void)n_in; (void)out_size; (void)ws_size;
  const float* x       = (const float*)d_in[0];
  const float* scale1  = (const float*)d_in[1];
  const float* offset1 = (const float*)d_in[2];
  const float* Wqkv    = (const float*)d_in[3];
  const float* bqkv    = (const float*)d_in[4];
  const float* a1      = (const float*)d_in[5];
  const float* a2      = (const float*)d_in[6];
  const float* bp1     = (const float*)d_in[7];
  const float* bp2     = (const float*)d_in[8];
  const float* Wout    = (const float*)d_in[9];
  const float* bout    = (const float*)d_in[10];
  const float* scale2  = (const float*)d_in[11];
  const float* offset2 = (const float*)d_in[12];
  const float* W1      = (const float*)d_in[13];
  const float* bW1     = (const float*)d_in[14];
  const float* W2      = (const float*)d_in[15];
  const float* bW2     = (const float*)d_in[16];
  const float* W3      = (const float*)d_in[17];
  const float* bW3     = (const float*)d_in[18];
  float* out = (float*)d_out;

  char* ws = (char*)d_ws;
  size_t off = 0;
  auto carve = [&](size_t bytes) -> char* {
    char* p = ws + off;
    off += (bytes + 255) & ~(size_t)255;
    return p;
  };
  unsigned short* wqkv_bf = (unsigned short*)carve((size_t)512 * 1536 * 2);
  unsigned short* wout_bf = (unsigned short*)carve((size_t)512 * 512 * 2);
  unsigned short* w1_bf   = (unsigned short*)carve((size_t)512 * 2048 * 2);
  unsigned short* w2_bf   = (unsigned short*)carve((size_t)512 * 2048 * 2);
  unsigned short* w3_bf   = (unsigned short*)carve((size_t)2048 * 512 * 2);
  float*          logbias = (float*)carve((size_t)288 * 4);
  unsigned short* buf512  = (unsigned short*)carve((size_t)M_TOT * 512 * 2);  // h -> ao -> g
  unsigned short* qkvbf   = (unsigned short*)carve((size_t)M_TOT * 1536 * 2);
  unsigned short* ffin    = (unsigned short*)carve((size_t)M_TOT * 2048 * 2);

  // 1) weights -> bf16
  convert_bf16_kernel<<<(512 * 1536 + 255) / 256, 256, 0, stream>>>(Wqkv, wqkv_bf, 512 * 1536);
  convert_bf16_kernel<<<(512 * 512 + 255) / 256, 256, 0, stream>>>(Wout, wout_bf, 512 * 512);
  convert_bf16_kernel<<<(512 * 2048 + 255) / 256, 256, 0, stream>>>(W1, w1_bf, 512 * 2048);
  convert_bf16_kernel<<<(512 * 2048 + 255) / 256, 256, 0, stream>>>(W2, w2_bf, 512 * 2048);
  convert_bf16_kernel<<<(2048 * 512 + 255) / 256, 256, 0, stream>>>(W3, w3_bf, 2048 * 512);

  // 2) decay-bias log table
  decay_kernel<<<1, 288, 0, stream>>>(a1, a2, bp1, bp2, logbias);

  // 3) RMSNorm1 -> h(bf16)
  rmsnorm_kernel<<<M_TOT, 256, 0, stream>>>(x, scale1, offset1, buf512);

  // 4) qkv = h @ Wqkv + bqkv  (bf16 out)
  gemm_bf16_kernel<0, 0><<<dim3(1536 / GBN, M_TOT / GBM), 256, 0, stream>>>(
      buf512, wqkv_bf, bqkv, nullptr, qkvbf, nullptr, M_TOT, 1536, 512);

  // 5) attention -> ao(bf16) (reuses buf512)
  attn_kernel<<<B_ * HEAD_ * 18, 288, ATTN_LDS_BYTES, stream>>>(qkvbf, logbias, buf512);

  // 6) x2 = ao @ Wout + bout + x  (fp32 -> d_out)
  gemm_bf16_kernel<1, 1><<<dim3(512 / GBN, M_TOT / GBM), 256, 0, stream>>>(
      buf512, wout_bf, bout, x, nullptr, out, M_TOT, 512, 512);

  // 7) RMSNorm2 -> g(bf16)
  rmsnorm_kernel<<<M_TOT, 256, 0, stream>>>(out, scale2, offset2, buf512);

  // 8) ffin = silu(g@W1+b1) * (g@W2+b2)  (fused dual GEMM)
  gemm_swiglu_kernel<<<dim3(2048 / GBN, M_TOT / GBM), 256, 0, stream>>>(
      buf512, w1_bf, bW1, w2_bf, bW2, ffin, M_TOT, 2048, 512);

  // 9) out = ffin @ W3 + bW3 + x2 (in-place residual on d_out)
  gemm_bf16_kernel<1, 1><<<dim3(512 / GBN, M_TOT / GBM), 256, 0, stream>>>(
      ffin, w3_bf, bW3, out, nullptr, out, M_TOT, 512, 2048);
}
